// GraphormerSelfAttention_52347061404287
// MI455X (gfx1250) — compile-verified, matrix operand order checked by reading, not run on hardware
//
#include <hip/hip_runtime.h>

#define B_   8
#define S_   1024
#define H_   768
#define NH_  12
#define DH_  64
#define ROWS (B_*S_)        // 8192
#define BH_  (B_*NH_)       // 96
#define QKSCALE 0.35355339059327373f  // 64^-0.25

typedef _Float16 half8 __attribute__((ext_vector_type(8)));
typedef _Float16 v16h  __attribute__((ext_vector_type(16)));
typedef float    v8f   __attribute__((ext_vector_type(8)));

__device__ __forceinline__ v16h cat16(half8 lo, half8 hi) {
    return __builtin_shufflevector(lo, hi, 0,1,2,3,4,5,6,7,8,9,10,11,12,13,14,15);
}

__device__ __forceinline__ v8f wmma_f16(v16h a, v16h b, v8f c) {
    return __builtin_amdgcn_wmma_f32_16x16x32_f16(false, a, false, b, (short)0, c, false, false);
}

// ---------------- 1. weight conversion f32 -> f16, [3][H][H] ----------------
__global__ void k_cvt_w(const float* __restrict__ Wq, const float* __restrict__ Wk,
                        const float* __restrict__ Wv, _Float16* __restrict__ wf16) {
    int idx = blockIdx.x * 256 + threadIdx.x;        // 0 .. 3*H*H-1
    const int HH = H_ * H_;
    int m = idx / HH;
    int r = idx - m * HH;
    const float* src = (m == 0) ? Wq : (m == 1) ? Wk : Wv;
    wf16[idx] = (_Float16)src[r];
}

// ---------------- 2. LayerNorm -> f16 ----------------
__global__ void k_ln(const float* __restrict__ x, const float* __restrict__ g,
                     const float* __restrict__ be, _Float16* __restrict__ xn) {
    const int row = blockIdx.x;
    const int t = threadIdx.x;
    const float* xr = x + (size_t)row * H_;
    float v0 = xr[t], v1 = xr[t + 256], v2 = xr[t + 512];
    __shared__ float r1[256], r2[256];
    r1[t] = v0 + v1 + v2;
    r2[t] = v0*v0 + v1*v1 + v2*v2;
    __syncthreads();
    for (int o = 128; o > 0; o >>= 1) {
        if (t < o) { r1[t] += r1[t + o]; r2[t] += r2[t + o]; }
        __syncthreads();
    }
    const float mu  = r1[0] * (1.0f / H_);
    const float var = r2[0] * (1.0f / H_) - mu * mu;
    const float ri  = rsqrtf(var + 1e-12f);
    _Float16* o = xn + (size_t)row * H_;
    o[t]       = (_Float16)((v0 - mu) * ri * g[t]       + be[t]);
    o[t + 256] = (_Float16)((v1 - mu) * ri * g[t + 256] + be[t + 256]);
    o[t + 512] = (_Float16)((v2 - mu) * ri * g[t + 512] + be[t + 512]);
}

// ---------------- 3. QKV GEMM: out = xn @ W^T + b (scaled), WMMA ----------------
// z = blockIdx.z selects q/k/v.  One 16x16 output tile per wave, K-loop 768/32.
__global__ void k_qkv(const _Float16* __restrict__ xn, const _Float16* __restrict__ wf16,
                      const float* __restrict__ bq, const float* __restrict__ bk,
                      const float* __restrict__ bv,
                      _Float16* __restrict__ qh, _Float16* __restrict__ kh,
                      _Float16* __restrict__ vt) {
    const int z    = blockIdx.z;
    const int wave = threadIdx.x >> 5;
    const int lane = threadIdx.x & 31;
    const int tid  = blockIdx.x * 8 + wave;      // 0..24575
    const int tm   = tid / (H_ / 16);            // 0..511
    const int tn   = tid % (H_ / 16);            // 0..47
    const int lr   = lane & 15;
    const int hi   = lane >> 4;
    const int k0   = hi ? 8 : 0;

    const _Float16* a_ptr = xn + (size_t)(tm * 16 + lr) * H_ + k0;
    const _Float16* b_ptr = wf16 + (size_t)z * H_ * H_ + (size_t)(tn * 16 + lr) * H_ + k0;

    v8f acc = {};
    for (int kb = 0; kb < H_; kb += 32) {
        half8 alo = *(const half8*)(a_ptr + kb);
        half8 ahi = *(const half8*)(a_ptr + kb + 16);
        half8 blo = *(const half8*)(b_ptr + kb);
        half8 bhi = *(const half8*)(b_ptr + kb + 16);
        acc = wmma_f16(cat16(alo, ahi), cat16(blo, bhi), acc);
    }

    const float* bias  = (z == 0) ? bq : (z == 1) ? bk : bv;
    const float  scale = (z == 2) ? 1.0f : QKSCALE;
    const int col  = tn * 16 + lr;
    const float bc = bias[col];
    const int head = col >> 6;       // 64 | 16, tile stays in one head
    const int d    = col & 63;
    #pragma unroll
    for (int r = 0; r < 8; ++r) {
        const int row  = tm * 16 + r + hi * 8;
        const int bb   = row >> 10;
        const int srow = row & 1023;
        const _Float16 val = (_Float16)((acc[r] + bc) * scale);
        if (z == 0)
            qh[(((size_t)bb * NH_ + head) * S_ + srow) * DH_ + d] = val;
        else if (z == 1)
            kh[(((size_t)bb * NH_ + head) * S_ + srow) * DH_ + d] = val;
        else
            vt[(((size_t)bb * NH_ + head) * DH_ + d) * S_ + srow] = val;
    }
}

// ---------------- 4. scores = q @ k^T + mask + bias (f32 to d_out) ----------------
__global__ void k_scores(const _Float16* __restrict__ qh, const _Float16* __restrict__ kh,
                         const float* __restrict__ mask, const float* __restrict__ abias,
                         float* __restrict__ scores) {
    const int bh   = blockIdx.y;                 // 0..95
    const int wave = threadIdx.x >> 5;
    const int lane = threadIdx.x & 31;
    const int tid  = blockIdx.x * 8 + wave;      // 0..4095
    const int tq   = tid >> 6;
    const int tk   = tid & 63;
    const int lr   = lane & 15;
    const int hi   = lane >> 4;
    const int k0   = hi ? 8 : 0;

    const _Float16* a_ptr = qh + ((size_t)bh * S_ + tq * 16 + lr) * DH_ + k0;
    const _Float16* b_ptr = kh + ((size_t)bh * S_ + tk * 16 + lr) * DH_ + k0;

    v8f acc = {};
    #pragma unroll
    for (int kb = 0; kb < DH_; kb += 32) {
        half8 alo = *(const half8*)(a_ptr + kb);
        half8 ahi = *(const half8*)(a_ptr + kb + 16);
        half8 blo = *(const half8*)(b_ptr + kb);
        half8 bhi = *(const half8*)(b_ptr + kb + 16);
        acc = wmma_f16(cat16(alo, ahi), cat16(blo, bhi), acc);
    }

    const int bidx = bh / NH_;
    const int kcol = tk * 16 + lr;
    const float m  = mask[(size_t)bidx * S_ + kcol];
    const float* brow = abias + (size_t)bh * S_ * S_;
    float* srow = scores + (size_t)bh * S_ * S_;
    #pragma unroll
    for (int r = 0; r < 8; ++r) {
        const int q = tq * 16 + r + hi * 8;
        srow[(size_t)q * S_ + kcol] = acc[r] + m + brow[(size_t)q * S_ + kcol];
    }
}

// ---------------- 5. softmax stats: row max and 1/sum(exp) ----------------
__global__ void k_stats(const float* __restrict__ scores,
                        float* __restrict__ rmax, float* __restrict__ rinv) {
    const int row = blockIdx.x;                  // 0 .. 96*1024-1
    const int t = threadIdx.x;
    const float* sr = scores + (size_t)row * S_;
    float v0 = sr[t], v1 = sr[t + 256], v2 = sr[t + 512], v3 = sr[t + 768];
    __shared__ float red[256];
    red[t] = fmaxf(fmaxf(v0, v1), fmaxf(v2, v3));
    __syncthreads();
    for (int o = 128; o > 0; o >>= 1) {
        if (t < o) red[t] = fmaxf(red[t], red[t + o]);
        __syncthreads();
    }
    const float mx = red[0];
    __syncthreads();
    red[t] = __expf(v0 - mx) + __expf(v1 - mx) + __expf(v2 - mx) + __expf(v3 - mx);
    __syncthreads();
    for (int o = 128; o > 0; o >>= 1) {
        if (t < o) red[t] += red[t + o];
        __syncthreads();
    }
    if (t == 0) { rmax[row] = mx; rinv[row] = 1.0f / red[0]; }
}

// ---------------- 6. ctx = softmax(scores) @ v  (probs rebuilt on the fly) -------
// One wave computes 16 q-rows x 64 d-cols (4 accumulators), K-loop 1024/32.
__global__ void k_ctx(const float* __restrict__ scores, const float* __restrict__ rmax,
                      const float* __restrict__ rinv, const _Float16* __restrict__ vt,
                      float* __restrict__ ctx) {
    const int bh   = blockIdx.y;                 // 0..95
    const int wave = threadIdx.x >> 5;
    const int lane = threadIdx.x & 31;
    const int tq   = blockIdx.x * 8 + wave;      // 0..63
    const int lr   = lane & 15;
    const int hi   = lane >> 4;
    const int k0   = hi ? 8 : 0;

    const int qrow = tq * 16 + lr;
    const float* srow = scores + ((size_t)bh * S_ + qrow) * S_;
    const float mx = rmax[bh * S_ + qrow];
    const float ri = rinv[bh * S_ + qrow];
    const _Float16* vbase = vt + (size_t)bh * DH_ * S_;

    v8f acc[4] = {};
    for (int kb = 0; kb < S_; kb += 32) {
        float pv[16];
        *(float4*)&pv[0]  = *(const float4*)(srow + kb + k0);
        *(float4*)&pv[4]  = *(const float4*)(srow + kb + k0 + 4);
        *(float4*)&pv[8]  = *(const float4*)(srow + kb + k0 + 16);
        *(float4*)&pv[12] = *(const float4*)(srow + kb + k0 + 20);
        v16h a;
        #pragma unroll
        for (int i = 0; i < 16; ++i)
            a[i] = (_Float16)(__expf(pv[i] - mx) * ri);
        #pragma unroll
        for (int dt = 0; dt < 4; ++dt) {
            const _Float16* bp = vbase + (size_t)(dt * 16 + lr) * S_ + kb + k0;
            half8 blo = *(const half8*)(bp);
            half8 bhi = *(const half8*)(bp + 16);
            acc[dt] = wmma_f16(a, cat16(blo, bhi), acc[dt]);
        }
    }

    const int bidx = bh / NH_;
    const int h    = bh % NH_;
    #pragma unroll
    for (int dt = 0; dt < 4; ++dt) {
        const int col = h * DH_ + dt * 16 + lr;
        #pragma unroll
        for (int r = 0; r < 8; ++r) {
            const int q = tq * 16 + r + hi * 8;
            ctx[((size_t)bidx * S_ + q) * H_ + col] = acc[dt][r];
        }
    }
}

extern "C" void kernel_launch(void* const* d_in, const int* in_sizes, int n_in,
                              void* d_out, int out_size, void* d_ws, size_t ws_size,
                              hipStream_t stream) {
    const float* hidden = (const float*)d_in[0];
    const float* mask   = (const float*)d_in[1];
    const float* abias  = (const float*)d_in[2];
    const float* gamma  = (const float*)d_in[3];
    const float* beta   = (const float*)d_in[4];
    const float* Wq = (const float*)d_in[5];  const float* bq = (const float*)d_in[6];
    const float* Wk = (const float*)d_in[7];  const float* bk = (const float*)d_in[8];
    const float* Wv = (const float*)d_in[9];  const float* bv = (const float*)d_in[10];

    float* ctx    = (float*)d_out;                          // [B,S,H]
    float* scores = (float*)d_out + (size_t)B_ * S_ * H_;   // [B,NH,S,S]

    char* ws = (char*)d_ws;
    _Float16* wf16 = (_Float16*)(ws + 0);            // 3*H*H f16      = 3,538,944 B
    _Float16* xn   = (_Float16*)(ws + 3538944);      // 8192*768 f16   = 12,582,912 B
    _Float16* qh   = (_Float16*)(ws + 16121856);     // [bh][s][d] f16
    _Float16* kh   = (_Float16*)(ws + 28704768);     // [bh][s][d] f16
    _Float16* vt   = (_Float16*)(ws + 41287680);     // [bh][d][s] f16
    float*    rmax = (float*)(ws + 53870592);        // 96*1024 f32
    float*    rinv = (float*)(ws + 54263808);        // 96*1024 f32

    k_cvt_w<<<3 * H_ * H_ / 256, 256, 0, stream>>>(Wq, Wk, Wv, wf16);
    k_ln<<<ROWS, 256, 0, stream>>>(hidden, gamma, beta, xn);
    {
        dim3 g((ROWS / 16) * (H_ / 16) / 8, 1, 3);   // (3072,1,3), 8 waves/block
        k_qkv<<<g, 256, 0, stream>>>(xn, wf16, bq, bk, bv, qh, kh, vt);
    }
    {
        dim3 g((64 * 64) / 8, BH_);                  // (512,96)
        k_scores<<<g, 256, 0, stream>>>(qh, kh, mask, abias, scores);
    }
    k_stats<<<BH_ * S_, 256, 0, stream>>>(scores, rmax, rinv);
    {
        dim3 g(64 / 8, BH_);                         // (8,96)
        k_ctx<<<g, 256, 0, stream>>>(scores, rmax, rinv, vt, ctx);
    }
}
